// FrameTransformer_73684458930615
// MI455X (gfx1250) — matrix operand order repaired and checked by reading, not compiled, not run on hardware
//
#include <hip/hip_runtime.h>
#include <math.h>

// ---------------- types / helpers ----------------
typedef __attribute__((ext_vector_type(16))) __bf16 v16bf;
typedef __attribute__((ext_vector_type(8)))  float  v8f;

__device__ __forceinline__ unsigned short f2bf(float f) {
    unsigned int u = __float_as_uint(f);
    unsigned int r = (u + 0x7FFFu + ((u >> 16) & 1u)) >> 16;   // RNE
    return (unsigned short)r;
}
__device__ __forceinline__ v16bf frag_from(uint4 a, uint4 b) {
    union { uint4 q[2]; v16bf v; } u;
    u.q[0] = a; u.q[1] = b;
    return u.v;
}
__device__ __forceinline__ float wred_sum(float v) {
    #pragma unroll
    for (int o = 16; o > 0; o >>= 1) v += __shfl_xor(v, o, 32);
    return v;
}
__device__ __forceinline__ float wred_max(float v) {
    #pragma unroll
    for (int o = 16; o > 0; o >>= 1) v = fmaxf(v, __shfl_xor(v, o, 32));
    return v;
}
// CDNA5 async global->LDS copy (ASYNCcnt tracked), per-lane 16B
__device__ __forceinline__ void async_ld_b128(unsigned int lds_off, const void* gaddr) {
    unsigned long long ga = (unsigned long long)(uintptr_t)gaddr;
    asm volatile("global_load_async_to_lds_b128 %0, %1, off" :: "v"(lds_off), "v"(ga) : "memory");
}
__device__ __forceinline__ void wait_async0() {
    asm volatile("s_wait_asynccnt 0x0" ::: "memory");
}
__device__ __forceinline__ unsigned int lds_off_of(const void* p) {
    return (unsigned int)(uintptr_t)p;   // LDS aperture: low 32 bits = LDS byte offset
}

// problem constants
constexpr int BATCH = 4, CIN = 8, CC = 4, FF = 1024, WW = 512, NHEADS = 8, DH = 128;

// ---------------- elementwise kernels ----------------
__global__ void cvt_f32_bf16(const float* __restrict__ in, unsigned short* __restrict__ out, size_t n) {
    size_t i = (size_t)blockIdx.x * 256 + threadIdx.x;
    if (i < n) out[i] = f2bf(in[i]);
}
__global__ void add_inplace(float* __restrict__ a, const float* __restrict__ b, size_t n) {
    size_t i = (size_t)blockIdx.x * 256 + threadIdx.x;
    if (i < n) a[i] += b[i];
}
__global__ void sqrelu_inplace(float* __restrict__ a, size_t n) {
    size_t i = (size_t)blockIdx.x * 256 + threadIdx.x;
    if (i < n) { float v = fmaxf(a[i], 0.0f); a[i] = v * v; }
}

// ---------------- 3x3 SAME conv (+bias), f32 (b,c,h,w) -> bf16 TRANSPOSED (b,c,w,h) -----
__global__ void conv3x3_bf16t(const float* __restrict__ in, const float* __restrict__ wgt,
                              const float* __restrict__ bias, unsigned short* __restrict__ out,
                              int Cin, int Cout, int H) {
    size_t idx = (size_t)blockIdx.x * 256 + threadIdx.x;
    int w = (int)(idx % WW); size_t t = idx / WW;
    int h = (int)(t % H); t /= H;
    int co = (int)(t % Cout); int b = (int)(t / Cout);
    float acc = bias[co];
    for (int ci = 0; ci < Cin; ++ci) {
        const float* ip = in + (size_t)(b * Cin + ci) * H * WW;
        const float* wp = wgt + (size_t)(co * Cin + ci) * 9;
        #pragma unroll
        for (int dy = -1; dy <= 1; ++dy) {
            int hh = h + dy; if (hh < 0 || hh >= H) continue;
            #pragma unroll
            for (int dx = -1; dx <= 1; ++dx) {
                int wi = w + dx; if (wi < 0 || wi >= WW) continue;
                acc += wp[(dy + 1) * 3 + (dx + 1)] * ip[(size_t)hh * WW + wi];
            }
        }
    }
    // N-major layout: row = w, col = h (feature contiguous) -> lane-contiguous WMMA B frags
    out[((size_t)(b * Cout + co) * WW + w) * H + h] = f2bf(acc);
}

// ---------------- batched WMMA GEMM:  Out[batch](MxN,f32) = A[batch%CH](MxK) @ Bt[batch](NxK)^T ----
// block: 256 thr (8 waves), tile 64x256, K step 32, double-buffered async global->LDS staging.
// Wave tile 32x64 (2x4 fragments): 12 ds_load_b128 per 8 WMMA.
__global__ void gemm_bf16(const unsigned short* __restrict__ A, const unsigned short* __restrict__ Bt,
                          float* __restrict__ Out, int M, int N, int K, int CH) {
    int batch = blockIdx.z;
    const unsigned short* Ab = A + (size_t)(batch % CH) * M * K;
    const unsigned short* Bb = Bt + (size_t)batch * K * N;   // (N x K) row-major
    float* Ob = Out + (size_t)batch * M * N;
    int m0 = blockIdx.y * 64;
    int n0 = blockIdx.x * 256;

    __shared__ __align__(16) unsigned short As[2][64 * 32];   // M x K tiles (K contiguous)
    __shared__ __align__(16) unsigned short Bs[2][256 * 32];  // N x K tiles (K contiguous)

    int tid = threadIdx.x;
    int wave = tid >> 5, lane = tid & 31;
    int wm = wave >> 2, wn = wave & 3;        // 2 (M) x 4 (N) wave grid
    int lrow = lane & 15, lhi = lane >> 4;

    v8f acc[2][4] = {};

    // stage(buf, k0): A = one b128/thread, B = four b128/thread (one column each)
    auto stage = [&](int buf, int k0) {
        int row = tid >> 2, segA = tid & 3;
        async_ld_b128(lds_off_of(&As[buf][row * 32 + segA * 8]),
                      Ab + (size_t)(m0 + row) * K + k0 + segA * 8);
        #pragma unroll
        for (int seg = 0; seg < 4; ++seg)
            async_ld_b128(lds_off_of(&Bs[buf][tid * 32 + seg * 8]),
                          Bb + (size_t)(n0 + tid) * K + k0 + seg * 8);
    };

    int buf = 0;
    stage(0, 0);
    for (int k0 = 0; k0 < K; k0 += 32) {
        wait_async0();          // this wave's issued copies complete
        __syncthreads();        // all waves' copies visible; previous buffer no longer read
        if (k0 + 32 < K) {
            stage(buf ^ 1, k0 + 32);          // overlap next DMA with this tile's WMMAs
            __builtin_prefetch((const void*)(Ab + (size_t)(m0 + (tid >> 2)) * K + k0 + 64), 0, 1);
        }
        v16bf afr[2], bfr[4];
        #pragma unroll
        for (int i = 0; i < 2; ++i) {          // A frag: lane-contiguous 2 x ds_load_b128
            const unsigned short* p = &As[buf][(wm * 32 + i * 16 + lrow) * 32 + lhi * 16];
            afr[i] = frag_from(*(const uint4*)p, *(const uint4*)(p + 8));
        }
        #pragma unroll
        for (int j = 0; j < 4; ++j) {          // B frag: lane-contiguous 2 x ds_load_b128
            const unsigned short* p = &Bs[buf][(wn * 64 + j * 16 + lrow) * 32 + lhi * 16];
            bfr[j] = frag_from(*(const uint4*)p, *(const uint4*)(p + 8));
        }
        #pragma unroll
        for (int i = 0; i < 2; ++i)
            #pragma unroll
            for (int j = 0; j < 4; ++j)
                acc[i][j] = __builtin_amdgcn_wmma_f32_16x16x32_bf16(
                    false, afr[i], false, bfr[j], (short)0, acc[i][j], false, false);
        buf ^= 1;
    }
    // D layout: VGPR r -> M = r + 8*(lane>=16); N = lane&15
    #pragma unroll
    for (int i = 0; i < 2; ++i)
        #pragma unroll
        for (int j = 0; j < 4; ++j) {
            int col = n0 + wn * 64 + j * 16 + lrow;
            int rowb = m0 + wm * 32 + i * 16 + lhi * 8;
            #pragma unroll
            for (int r = 0; r < 8; ++r)
                Ob[(size_t)(rowb + r) * N + col] = acc[i][j][r];
        }
}

// ---------------- per-channel LayerNorm over feature axis, affine --------------
__global__ void ln_kernel(const float* __restrict__ in, const float* __restrict__ g,
                          const float* __restrict__ bta, float* __restrict__ out) {
    int wave = threadIdx.x >> 5, lane = threadIdx.x & 31;
    int col = blockIdx.x * 8 + wave;            // (bc, w) column
    int bc = col / WW, w = col % WW;
    int c = bc % CC;
    const float* ip = in + (size_t)bc * FF * WW + w;
    float xs[32]; float sum = 0.f, sq = 0.f;
    #pragma unroll
    for (int i = 0; i < 32; ++i) {
        float v = ip[(size_t)(lane + 32 * i) * WW];
        xs[i] = v; sum += v; sq += v * v;
    }
    sum = wred_sum(sum); sq = wred_sum(sq);
    float mu = sum * (1.0f / FF);
    float var = sq * (1.0f / FF) - mu * mu;
    float rs = rsqrtf(var + 1e-5f);
    float* op = out + (size_t)bc * FF * WW + w;
    #pragma unroll
    for (int i = 0; i < 32; ++i) {
        int f = lane + 32 * i;
        op[(size_t)f * WW] = (xs[i] - mu) * rs * g[c * FF + f] + bta[c * FF + f];
    }
}

// ---------------- RoPE (interleaved pairs, pos = w) f32 (bc,f,w) -> bf16 TRANSPOSED (bc,w,f) ----
__global__ void rope_bf16t(const float* __restrict__ in, unsigned short* __restrict__ out) {
    size_t idx = (size_t)blockIdx.x * 256 + threadIdx.x;     // one pair per thread
    int w = (int)(idx % WW); size_t t = idx / WW;
    int fp = (int)(t % (FF / 2)); int bc = (int)(t / (FF / 2));
    int i = fp & 63;                      // ((2*fp) % DH)/2
    float ang = (float)w * __expf(-(float)i * 0.14391156510954664f); // ln(1e4)/64
    float s, c; __sincosf(ang, &s, &c);
    size_t base = ((size_t)bc * FF + 2 * fp) * WW + w;
    float x0 = in[base], x1 = in[base + WW];
    unsigned int lo = f2bf(x0 * c - x1 * s);
    unsigned int hi = f2bf(x1 * c + x0 * s);
    // transposed: row = w, feature contiguous; pair -> one dword store
    ((unsigned int*)out)[(((size_t)bc * WW + w) * FF >> 1) + fp] = (hi << 16) | lo;
}

// ---------------- attention: per (b,c,head, 32-query block) ----------------
// qt/kt are (bc, w, f) bf16 ; vb is (bc, f, w) bf16 ; out f32 (bc, f, w)
__global__ void attn_kernel(const unsigned short* __restrict__ qt, const unsigned short* __restrict__ kt,
                            const unsigned short* __restrict__ vb, float* __restrict__ out) {
    int q0 = blockIdx.x * 32;
    int head = blockIdx.y;
    int bc = blockIdx.z;
    int hoff = head * DH;
    size_t baseF = (size_t)bc * FF * WW + (size_t)hoff * WW; // for (bc,f,w) tensors

    __shared__ float          S[32 * 512];        // 64 KB scores
    __shared__ unsigned short P[32 * 512];        // 32 KB bf16 probs
    __shared__ __align__(16) unsigned short Qs[32 * DH]; // 8 KB q tile

    int tid = threadIdx.x, wave = tid >> 5, lane = tid & 31;
    int lrow = lane & 15, lhi = lane >> 4;
    const float scale = 1.0f / 32.0f;             // 1/sqrt(1024)

    // stage q tile (32 queries x 128 dims) via async DMA, 2 x b128 per thread
    {
        int row = tid >> 3, seg = tid & 7;
        const unsigned short* src = qt + ((size_t)(bc * WW + q0 + row)) * FF + hoff + seg * 16;
        async_ld_b128(lds_off_of(Qs + row * DH + seg * 16), src);
        async_ld_b128(lds_off_of(Qs + row * DH + seg * 16 + 8), src + 8);
    }
    wait_async0();
    __syncthreads();

    // Phase 1: S = (q @ k^T) * scale ; 8 waves tile keys, 2 passes of 256
    for (int nc = 0; nc < 2; ++nc) {
        int n_off = nc * 256 + wave * 32;
        v8f acc[2][2] = {};
        for (int kd = 0; kd < DH; kd += 32) {
            v16bf afr[2], bfr[2];
            #pragma unroll
            for (int i = 0; i < 2; ++i) {      // q frag from LDS: 2 x ds_load_b128
                const unsigned short* p = Qs + (i * 16 + lrow) * DH + kd + lhi * 16;
                afr[i] = frag_from(*(const uint4*)p, *(const uint4*)(p + 8));
            }
            #pragma unroll
            for (int jj = 0; jj < 2; ++jj) {   // k frag from global: 2 x global_load_b128
                const unsigned short* p = kt + ((size_t)(bc * WW + n_off + jj * 16 + lrow)) * FF
                                             + hoff + kd + lhi * 16;
                bfr[jj] = frag_from(*(const uint4*)p, *(const uint4*)(p + 8));
            }
            #pragma unroll
            for (int i = 0; i < 2; ++i)
                #pragma unroll
                for (int jj = 0; jj < 2; ++jj)
                    acc[i][jj] = __builtin_amdgcn_wmma_f32_16x16x32_bf16(
                        false, afr[i], false, bfr[jj], (short)0, acc[i][jj], false, false);
        }
        #pragma unroll
        for (int i = 0; i < 2; ++i)
            #pragma unroll
            for (int jj = 0; jj < 2; ++jj) {
                int colS = n_off + jj * 16 + lrow;
                int rowS = i * 16 + lhi * 8;
                #pragma unroll
                for (int r = 0; r < 8; ++r)
                    S[(rowS + r) * 512 + colS] = acc[i][jj][r] * scale;
            }
    }
    __syncthreads();

    // Phase 2: row softmax -> bf16 probabilities in P
    for (int rr = 0; rr < 4; ++rr) {
        int row = wave * 4 + rr;
        float mx = -1e30f;
        #pragma unroll
        for (int ii = 0; ii < 16; ++ii) mx = fmaxf(mx, S[row * 512 + lane + ii * 32]);
        mx = wred_max(mx);
        float sum = 0.f; float e[16];
        #pragma unroll
        for (int ii = 0; ii < 16; ++ii) {
            e[ii] = __expf(S[row * 512 + lane + ii * 32] - mx);
            sum += e[ii];
        }
        sum = wred_sum(sum);
        float inv = 1.0f / sum;
        #pragma unroll
        for (int ii = 0; ii < 16; ++ii) P[row * 512 + lane + ii * 32] = f2bf(e[ii] * inv);
    }
    __syncthreads();

    // Phase 3: O = P @ V ; waves 2(M=16) x 4(N=32)
    int wm = wave >> 2, wn = wave & 3;
    v8f acc2[2] = {};
    for (int kk = 0; kk < 512; kk += 32) {
        v16bf afr, bfr[2];
        {   // P frag from LDS: 2 x ds_load_b128
            const unsigned short* p = P + (wm * 16 + lrow) * 512 + kk + lhi * 16;
            afr = frag_from(*(const uint4*)p, *(const uint4*)(p + 8));
        }
        #pragma unroll
        for (int jj = 0; jj < 2; ++jj) {       // v frag from global: 2 x global_load_b128
            const unsigned short* p = vb + baseF + (size_t)(wn * 32 + jj * 16 + lrow) * WW
                                         + kk + lhi * 16;
            bfr[jj] = frag_from(*(const uint4*)p, *(const uint4*)(p + 8));
        }
        #pragma unroll
        for (int jj = 0; jj < 2; ++jj)
            acc2[jj] = __builtin_amdgcn_wmma_f32_16x16x32_bf16(
                false, afr, false, bfr[jj], (short)0, acc2[jj], false, false);
    }
    #pragma unroll
    for (int jj = 0; jj < 2; ++jj) {
        int d = wn * 32 + jj * 16 + lrow;
        #pragma unroll
        for (int r = 0; r < 8; ++r)
            out[baseF + (size_t)d * WW + q0 + wm * 16 + r + 8 * lhi] = acc2[jj][r];
    }
}

// ---------------- final concat output ----------------
__global__ void assemble_out(const float* __restrict__ x, const float* __restrict__ h,
                             float* __restrict__ out) {
    size_t idx = (size_t)blockIdx.x * 256 + threadIdx.x;
    int w = (int)(idx % WW); size_t t = idx / WW;
    int f = (int)(t % FF); t /= FF;
    int ch = (int)(t % (CIN + CC)); int b = (int)(t / (CIN + CC));
    float v;
    if (ch < CIN) v = x[(((size_t)b * CIN + ch) * FF + f) * WW + w];
    else          v = h[(((size_t)b * CC + (ch - CIN)) * FF + f) * WW + w];
    out[idx] = v;
}

// ---------------- host-side launch ----------------
extern "C" void kernel_launch(void* const* d_in, const int* in_sizes, int n_in,
                              void* d_out, int out_size, void* d_ws, size_t ws_size,
                              hipStream_t stream) {
    (void)in_sizes; (void)n_in; (void)out_size; (void)ws_size;
    const float* x        = (const float*)d_in[0];
    const float* embed_cw = (const float*)d_in[1];
    const float* embed_cb = (const float*)d_in[2];
    const float* embed_pw = (const float*)d_in[3];
    const float* n1g      = (const float*)d_in[4];
    const float* n1b      = (const float*)d_in[5];
    const float* q_cw = (const float*)d_in[6];  const float* q_cb = (const float*)d_in[7];  const float* q_pw = (const float*)d_in[8];
    const float* k_cw = (const float*)d_in[9];  const float* k_cb = (const float*)d_in[10]; const float* k_pw = (const float*)d_in[11];
    const float* v_cw = (const float*)d_in[12]; const float* v_cb = (const float*)d_in[13]; const float* v_pw = (const float*)d_in[14];
    const float* o_cw = (const float*)d_in[15]; const float* o_cb = (const float*)d_in[16]; const float* o_pw = (const float*)d_in[17];
    const float* n2g  = (const float*)d_in[18]; const float* n2b  = (const float*)d_in[19];
    const float* f1cw = (const float*)d_in[20]; const float* f1cb = (const float*)d_in[21]; const float* f1pw = (const float*)d_in[22];
    const float* f2cw = (const float*)d_in[23]; const float* f2cb = (const float*)d_in[24]; const float* f2pw = (const float*)d_in[25];

    char* ws = (char*)d_ws;
    const size_t MB = 1024ull * 1024ull;
    float*          h0   = (float*)(ws + 0);                  // 32 MB
    float*          tA   = (float*)(ws + 32 * MB);            // 32 MB
    float*          tB   = (float*)(ws + 64 * MB);            // 32 MB
    unsigned short* cb16 = (unsigned short*)(ws + 96 * MB);   // 16 MB (N-major conv out)
    unsigned short* qb16 = (unsigned short*)(ws + 112 * MB);  // 16 MB (w-major)
    unsigned short* kb16 = (unsigned short*)(ws + 128 * MB);  // 16 MB (w-major)
    unsigned short* vb16 = (unsigned short*)(ws + 144 * MB);  // 16 MB (f-major)
    float*          hidf = (float*)(ws + 160 * MB);           // 128 MB
    unsigned short* hidb = (unsigned short*)(ws + 288 * MB);  // 64 MB (N-major)
    unsigned short* wemb = (unsigned short*)(ws + 352 * MB);  // 8 MB each
    unsigned short* wq   = (unsigned short*)(ws + 360 * MB);
    unsigned short* wk   = (unsigned short*)(ws + 368 * MB);
    unsigned short* wv   = (unsigned short*)(ws + 376 * MB);
    unsigned short* wo   = (unsigned short*)(ws + 384 * MB);
    unsigned short* wf1  = (unsigned short*)(ws + 392 * MB);  // 32 MB
    unsigned short* wf2  = (unsigned short*)(ws + 424 * MB);  // 32 MB

    const size_t T    = (size_t)BATCH * CC * FF * WW;   // 8,388,608
    const size_t T4   = T * 4;
    const size_t WSML = (size_t)CC * FF * FF;           // 4,194,304
    const size_t WBIG = (size_t)CC * FF * FF * 4;       // 16,777,216
    dim3 blk(256);

    // weight downconvert f32 -> bf16
    cvt_f32_bf16<<<(WSML + 255) / 256, blk, 0, stream>>>(embed_pw, wemb, WSML);
    cvt_f32_bf16<<<(WSML + 255) / 256, blk, 0, stream>>>(q_pw,  wq,  WSML);
    cvt_f32_bf16<<<(WSML + 255) / 256, blk, 0, stream>>>(k_pw,  wk,  WSML);
    cvt_f32_bf16<<<(WSML + 255) / 256, blk, 0, stream>>>(v_pw,  wv,  WSML);
    cvt_f32_bf16<<<(WSML + 255) / 256, blk, 0, stream>>>(o_pw,  wo,  WSML);
    cvt_f32_bf16<<<(WBIG + 255) / 256, blk, 0, stream>>>(f1pw, wf1, WBIG);
    cvt_f32_bf16<<<(WBIG + 255) / 256, blk, 0, stream>>>(f2pw, wf2, WBIG);

    // embed MCL: conv + GEMM -> h0
    conv3x3_bf16t<<<(unsigned)(T / 256), blk, 0, stream>>>(x, embed_cw, embed_cb, cb16, CIN, CC, FF);
    gemm_bf16<<<dim3(2, 16, 16), blk, 0, stream>>>(wemb, cb16, h0, FF, WW, FF, CC);

    // attention branch
    ln_kernel<<<1024, blk, 0, stream>>>(h0, n1g, n1b, tA);

    conv3x3_bf16t<<<(unsigned)(T / 256), blk, 0, stream>>>(tA, q_cw, q_cb, cb16, CC, CC, FF);
    gemm_bf16<<<dim3(2, 16, 16), blk, 0, stream>>>(wq, cb16, tB, FF, WW, FF, CC);
    rope_bf16t<<<(unsigned)(T / 2 / 256), blk, 0, stream>>>(tB, qb16);

    conv3x3_bf16t<<<(unsigned)(T / 256), blk, 0, stream>>>(tA, k_cw, k_cb, cb16, CC, CC, FF);
    gemm_bf16<<<dim3(2, 16, 16), blk, 0, stream>>>(wk, cb16, tB, FF, WW, FF, CC);
    rope_bf16t<<<(unsigned)(T / 2 / 256), blk, 0, stream>>>(tB, kb16);

    conv3x3_bf16t<<<(unsigned)(T / 256), blk, 0, stream>>>(tA, v_cw, v_cb, cb16, CC, CC, FF);
    gemm_bf16<<<dim3(2, 16, 16), blk, 0, stream>>>(wv, cb16, tB, FF, WW, FF, CC);
    cvt_f32_bf16<<<(unsigned)(T / 256), blk, 0, stream>>>(tB, vb16, T);   // keep f-major

    attn_kernel<<<dim3(16, NHEADS, 16), blk, 0, stream>>>(qb16, kb16, vb16, tB);

    conv3x3_bf16t<<<(unsigned)(T / 256), blk, 0, stream>>>(tB, o_cw, o_cb, cb16, CC, CC, FF);
    gemm_bf16<<<dim3(2, 16, 16), blk, 0, stream>>>(wo, cb16, tA, FF, WW, FF, CC);
    add_inplace<<<(unsigned)(T / 256), blk, 0, stream>>>(h0, tA, T);

    // FFN branch
    ln_kernel<<<1024, blk, 0, stream>>>(h0, n2g, n2b, tA);
    conv3x3_bf16t<<<(unsigned)(T / 256), blk, 0, stream>>>(tA, f1cw, f1cb, cb16, CC, CC, FF);
    gemm_bf16<<<dim3(2, 64, 16), blk, 0, stream>>>(wf1, cb16, hidf, FF * 4, WW, FF, CC);
    sqrelu_inplace<<<(unsigned)(T4 / 256), blk, 0, stream>>>(hidf, T4);
    conv3x3_bf16t<<<(unsigned)(T4 / 256), blk, 0, stream>>>(hidf, f2cw, f2cb, hidb, CC, CC, FF * 4);
    gemm_bf16<<<dim3(2, 16, 16), blk, 0, stream>>>(wf2, hidb, tA, FF, WW, FF * 4, CC);
    add_inplace<<<(unsigned)(T / 256), blk, 0, stream>>>(h0, tA, T);

    // concat [x, h] along channels
    size_t outElems = (size_t)BATCH * (CIN + CC) * FF * WW;   // 25,165,824
    assemble_out<<<(unsigned)(outElems / 256), blk, 0, stream>>>(x, h0, (float*)d_out);
}
